// ISTFT_66503273612036
// MI455X (gfx1250) — matrix operand order repaired and checked by reading, not compiled, not run on hardware
//
#include <hip/hip_runtime.h>
#include <math.h>

typedef __attribute__((ext_vector_type(16))) _Float16 v16h;
typedef __attribute__((ext_vector_type(8)))  float    v8f;
typedef __attribute__((ext_vector_type(4)))  float    f32x4;
typedef __attribute__((ext_vector_type(4)))  int      i32x4;

typedef __attribute__((address_space(1))) i32x4 gi32x4;   // global int4
typedef __attribute__((address_space(3))) i32x4 li32x4;   // LDS int4

#define BATCH      16
#define T_FRAMES   601
#define F_BINS     1025
#define NFFT       2048
#define HOP        512
#define OUT_LEN    307200
#define EXPECT_LEN 309248            // NFFT + HOP*(T_FRAMES-1)
#define KDIM       2080              // 2*F_BINS = 2050 padded to 65*32
#define KITER      (KDIM / 32)       // 65
#define NCOLS      (BATCH*T_FRAMES)  // 9616
#define NPAD       9728              // 76*128
#define A_SCALE    512.0f
#define INV_SCALE  (1.0f/512.0f)
#define PI_F       3.14159265358979323846f
#define TINY_F32   1.17549435e-38f

// ---- workspace layout (bytes) ----
#define WS_Y_BYTES ((size_t)BATCH * EXPECT_LEN * 4)            // fp32 accum
#define WS_A_OFF   (WS_Y_BYTES)                                 // f16 A [2048][KDIM]
#define WS_A_BYTES ((size_t)2048 * KDIM * 2)
#define WS_B_OFF   (WS_A_OFF + WS_A_BYTES)                      // f16 B^T [NPAD][KDIM]

#if __has_builtin(__builtin_amdgcn_global_load_async_to_lds_b128) && \
    __has_builtin(__builtin_amdgcn_s_wait_asynccnt)
#define USE_ASYNC_LDS 1
#else
#define USE_ASYNC_LDS 0
#endif

// per-buffer LDS layout (halves): A tile [0,4096) rows 128 x 32, B tile [4096,8192)
#define LDS_BUF_H  8192
#define LDS_B_OFF  4096

// -------------------------------------------------------------------------
__global__ void zero_f32(float* __restrict__ p, int n) {
    int i = blockIdx.x * blockDim.x + threadIdx.x;
    if (i < n) p[i] = 0.0f;
}

// Fold Hermitian symmetry into the IDFT weights; scale by A_SCALE; store f16.
__global__ void pack_A(const float* __restrict__ wr, const float* __restrict__ wi,
                       _Float16* __restrict__ A) {
    int idx = blockIdx.x * blockDim.x + threadIdx.x;
    if (idx >= 2048 * KDIM) return;
    int o = idx / KDIM;
    int k = idx - o * KDIM;
    float v = 0.0f;
    if (k < F_BINS) {
        int f = k;
        v = wr[o * 2048 + f];
        if (f >= 1 && f <= 1023) v += wr[o * 2048 + (2048 - f)];
    } else if (k < 2 * F_BINS) {
        int f = k - F_BINS;
        v = -wi[o * 2048 + f];
        if (f >= 1 && f <= 1023) v += wi[o * 2048 + (2048 - f)];
    }
    A[idx] = (_Float16)(v * A_SCALE);
}

// B^T[n][k]: n = b*T + t (input rows are contiguous in f, so this is a plain
// f32->f16 convert with zero padding).
__global__ void pack_B(const float* __restrict__ re, const float* __restrict__ im,
                       _Float16* __restrict__ Bt) {
    int idx = blockIdx.x * blockDim.x + threadIdx.x;
    if (idx >= NPAD * KDIM) return;
    int n = idx / KDIM;
    int k = idx - n * KDIM;
    float v = 0.0f;
    if (n < NCOLS) {
        if (k < F_BINS)            v = re[(size_t)n * F_BINS + k];
        else if (k < 2 * F_BINS)   v = im[(size_t)n * F_BINS + (k - F_BINS)];
    }
    Bt[idx] = (_Float16)v;
}

// -------------------------------------------------------------------------
// Tiled WMMA GEMM with double-buffered async global->LDS staging.
// Block tile 128(M=o) x 128(N=n); 8 waves in 2(M) x 4(N); wave tile 64x32
// = 8 accumulators. K-step 32, 65 steps.
__global__ __launch_bounds__(256) void istft_gemm(
        const _Float16* __restrict__ A,   // [2048][KDIM]
        const _Float16* __restrict__ Bt,  // [NPAD][KDIM]
        float* __restrict__ y)            // [BATCH][EXPECT_LEN]
{
    __shared__ __align__(16) _Float16 lds[2 * LDS_BUF_H]; // 32 KB

    const int tid    = threadIdx.x;
    const int lane   = tid & 31;
    const int wave   = tid >> 5;
    const int wm     = wave & 1;       // 0..1 -> 64-row M slice
    const int wn     = wave >> 1;      // 0..3 -> 32-col N slice
    const int laneHi = lane >> 4;
    const int lane16 = lane & 15;
    const int mBase  = blockIdx.x * 128;
    const int nBase  = blockIdx.y * 128;

    // staging: 16B chunks; each tile (A or B) = 128 rows x 4 chunks = 512
    // chunks; each thread owns rows r and r+64 at column chunk c.
    const int sRow = tid >> 2;             // 0..63
    const int sCol = tid & 3;              // 0..3

    const _Float16* gA0 = A  + (size_t)(mBase + sRow)      * KDIM + sCol * 8;
    const _Float16* gA1 = A  + (size_t)(mBase + sRow + 64) * KDIM + sCol * 8;
    const _Float16* gB0 = Bt + (size_t)(nBase + sRow)      * KDIM + sCol * 8;
    const _Float16* gB1 = Bt + (size_t)(nBase + sRow + 64) * KDIM + sCol * 8;

    const int ldsA0 = sRow * 32 + sCol * 8;
    const int ldsA1 = (sRow + 64) * 32 + sCol * 8;
    const int ldsB0 = LDS_B_OFF + sRow * 32 + sCol * 8;
    const int ldsB1 = LDS_B_OFF + (sRow + 64) * 32 + sCol * 8;

#if USE_ASYNC_LDS
#define ISSUE_TILE(bufbase, kk)                                                  \
    do {                                                                         \
        __builtin_amdgcn_global_load_async_to_lds_b128(                          \
            (gi32x4*)(gA0 + (kk)), (li32x4*)&lds[(bufbase) + ldsA0], 0, 0);      \
        __builtin_amdgcn_global_load_async_to_lds_b128(                          \
            (gi32x4*)(gA1 + (kk)), (li32x4*)&lds[(bufbase) + ldsA1], 0, 0);      \
        __builtin_amdgcn_global_load_async_to_lds_b128(                          \
            (gi32x4*)(gB0 + (kk)), (li32x4*)&lds[(bufbase) + ldsB0], 0, 0);      \
        __builtin_amdgcn_global_load_async_to_lds_b128(                          \
            (gi32x4*)(gB1 + (kk)), (li32x4*)&lds[(bufbase) + ldsB1], 0, 0);      \
    } while (0)
#else
#define ISSUE_TILE(bufbase, kk)                                                  \
    do {                                                                         \
        *(f32x4*)&lds[(bufbase) + ldsA0] = *(const f32x4*)(gA0 + (kk));          \
        *(f32x4*)&lds[(bufbase) + ldsA1] = *(const f32x4*)(gA1 + (kk));          \
        *(f32x4*)&lds[(bufbase) + ldsB0] = *(const f32x4*)(gB0 + (kk));          \
        *(f32x4*)&lds[(bufbase) + ldsB1] = *(const f32x4*)(gB1 + (kk));          \
    } while (0)
#endif

    v8f acc[8] = {};   // acc[i*2+j]: i = M sub-tile 0..3, j = N sub-tile 0..1

    ISSUE_TILE(0, 0);

#pragma unroll 1
    for (int it = 0; it < KITER; ++it) {
        const int curBase = (it & 1) ? LDS_BUF_H : 0;
        if (it + 1 < KITER) {
            const int nxtBase = (it & 1) ? 0 : LDS_BUF_H;
            const int kk = (it + 1) * 32;
            __builtin_prefetch(gA0 + kk + 32, 0, 3);   // global_prefetch_b8
            __builtin_prefetch(gB0 + kk + 32, 0, 3);
            ISSUE_TILE(nxtBase, kk);
#if USE_ASYNC_LDS
            __builtin_amdgcn_s_wait_asynccnt(4);   // tile `it` done, next in flight
#endif
        } else {
#if USE_ASYNC_LDS
            __builtin_amdgcn_s_wait_asynccnt(0);
#endif
        }
        __syncthreads();   // all waves' staging of buf `cur` visible

        union HV { v16h v; f32x4 q[2]; };

        // B fragments (32x16 KxN): lane holds col N=lane16; lanes 0-15 carry
        // K {0..7,16..23}, lanes 16-31 carry K {8..15,24..31}.
        HV bf[2];
#pragma unroll
        for (int j = 0; j < 2; ++j) {
            const f32x4* pb = (const f32x4*)
                &lds[curBase + LDS_B_OFF + (wn * 32 + j * 16 + lane16) * 32 + laneHi * 8];
            bf[j].q[0] = pb[0];
            bf[j].q[1] = pb[2];
        }

        // A fragments (16x32 MxK): lane holds row M=lane16; lanes 0-15 carry
        // K {0..7,16..23}, lanes 16-31 carry K {8..15,24..31}.
#pragma unroll
        for (int i = 0; i < 4; ++i) {
            HV af;
            const _Float16* pa =
                &lds[curBase + (wm * 64 + i * 16 + lane16) * 32 + laneHi * 8];
            af.q[0] = *(const f32x4*)pa;
            af.q[1] = *(const f32x4*)(pa + 16);
#pragma unroll
            for (int j = 0; j < 2; ++j) {
                acc[i * 2 + j] = __builtin_amdgcn_wmma_f32_16x16x32_f16(
                    false, af.v, false, bf[j].v, (short)0, acc[i * 2 + j], false, false);
            }
        }
        __syncthreads();   // buf `cur` free for reuse next iteration
    }

    // Epilogue: C layout -> VGPR e, lanes 0-15: M=e, lanes 16-31: M=e+8; N=lane16.
    // frames[b,t,o] scatter-adds into y[b][t*HOP + o].
#pragma unroll
    for (int j = 0; j < 2; ++j) {
        const int nGlob = nBase + wn * 32 + j * 16 + lane16;
        if (nGlob < NCOLS) {
            const int bIdx = nGlob / T_FRAMES;
            const int tIdx = nGlob - bIdx * T_FRAMES;
            float* yb = y + (size_t)bIdx * EXPECT_LEN + (size_t)tIdx * HOP;
#pragma unroll
            for (int i = 0; i < 4; ++i) {
                const int oBase = mBase + wm * 64 + i * 16 + laneHi * 8;
#pragma unroll
                for (int e = 0; e < 8; ++e) {
                    __hip_atomic_fetch_add(&yb[oBase + e], acc[i * 2 + j][e] * INV_SCALE,
                                           __ATOMIC_RELAXED, __HIP_MEMORY_SCOPE_AGENT);
                }
            }
        }
    }
#undef ISSUE_TILE
}

// -------------------------------------------------------------------------
// Window-sumsquare normalization + center trim. hann(n)=sin^2(pi*n/NFFT),
// so win_sq(n)=sin^4. Each sample p has <=4 contributing frames.
__global__ void finalize(const float* __restrict__ y, float* __restrict__ out) {
    int idx = blockIdx.x * blockDim.x + threadIdx.x;
    if (idx >= BATCH * OUT_LEN) return;
    int b = idx / OUT_LEN;
    int l = idx - b * OUT_LEN;
    int p = l + NFFT / 2;
    float acc = y[(size_t)b * EXPECT_LEN + p];
    int tTop = p >> 9;
    int nOff = p & 511;
    float ws = 0.0f;
#pragma unroll
    for (int j = 0; j < 4; ++j) {
        int t = tTop - j;
        if (t >= 0 && t < T_FRAMES) {
            float s  = __sinf(PI_F * (float)(nOff + 512 * j) * (1.0f / (float)NFFT));
            float s2 = s * s;
            ws += s2 * s2;
        }
    }
    out[idx] = (ws > TINY_F32) ? (acc / ws) : acc;
}

// -------------------------------------------------------------------------
extern "C" void kernel_launch(void* const* d_in, const int* in_sizes, int n_in,
                              void* d_out, int out_size, void* d_ws, size_t ws_size,
                              hipStream_t stream) {
    (void)in_sizes; (void)n_in; (void)out_size; (void)ws_size;
    const float* re = (const float*)d_in[0];  // (16,1,601,1025)
    const float* im = (const float*)d_in[1];  // (16,1,601,1025)
    const float* wr = (const float*)d_in[2];  // (2048,2048)
    const float* wi = (const float*)d_in[3];  // (2048,2048)
    float* out = (float*)d_out;               // (16,307200)

    char* ws = (char*)d_ws;
    float*    y  = (float*)ws;
    _Float16* Aw = (_Float16*)(ws + WS_A_OFF);
    _Float16* Bw = (_Float16*)(ws + WS_B_OFF);

    {   int n = BATCH * EXPECT_LEN;
        zero_f32<<<(n + 255) / 256, 256, 0, stream>>>(y, n); }
    {   int n = 2048 * KDIM;
        pack_A<<<(n + 255) / 256, 256, 0, stream>>>(wr, wi, Aw); }
    {   int n = NPAD * KDIM;
        pack_B<<<(n + 255) / 256, 256, 0, stream>>>(re, im, Bw); }

    istft_gemm<<<dim3(2048 / 128, NPAD / 128), 256, 0, stream>>>(Aw, Bw, y);

    {   int n = BATCH * OUT_LEN;
        finalize<<<(n + 255) / 256, 256, 0, stream>>>(y, out); }
}
